// dynamic_reg_85323820302424
// MI455X (gfx1250) — matrix operand order repaired and checked by reading, not compiled, not run on hardware
//
#include <hip/hip_runtime.h>
#include <stdint.h>

// Problem constants (from reference)
#define NN   2048
#define BB   16
#define TT   12
#define HIDD 64
#define ED   10
#define HORZ 12

typedef __attribute__((ext_vector_type(16))) __bf16 v16b;
typedef __attribute__((ext_vector_type(8)))  __bf16 v8b;
typedef __attribute__((ext_vector_type(8)))  float  v8f;

static __device__ __forceinline__ v16b mk16(v8b lo, v8b hi) {
  return __builtin_shufflevector(lo, hi, 0,1,2,3,4,5,6,7,8,9,10,11,12,13,14,15);
}
static __device__ __forceinline__ v8f wmma_bf16(v16b a, v16b b, v8f c) {
  // D = A(16x32 bf16) x B(32x16 bf16) + C(16x16 f32)
  return __builtin_amdgcn_wmma_f32_16x16x32_bf16(false, a, false, b, (short)0, c, false, false);
}

// ---------------------------------------------------------------------------
// S = softmax(relu(E E^T), axis=1), stored bf16 row-major [2048][2048]
// one block per row
// ---------------------------------------------------------------------------
__global__ __launch_bounds__(256) void k_supports(const float* __restrict__ E,
                                                  __bf16* __restrict__ S) {
  __shared__ float rowv[NN];
  __shared__ float red[256];
  const int row = blockIdx.x, tid = threadIdx.x;
  float er[ED];
#pragma unroll
  for (int d = 0; d < ED; ++d) er[d] = E[row * ED + d];
  float lm = 0.f;
  for (int m = tid; m < NN; m += 256) {
    const float* em = E + m * ED;
    float dot = 0.f;
#pragma unroll
    for (int d = 0; d < ED; ++d) dot += er[d] * em[d];
    float v = dot > 0.f ? dot : 0.f;   // relu
    rowv[m] = v;
    lm = fmaxf(lm, v);
  }
  red[tid] = lm; __syncthreads();
  for (int s = 128; s > 0; s >>= 1) {
    if (tid < s) red[tid] = fmaxf(red[tid], red[tid + s]);
    __syncthreads();
  }
  const float gmax = red[0]; __syncthreads();
  float ls = 0.f;
  for (int m = tid; m < NN; m += 256) {
    float e = __expf(rowv[m] - gmax);
    rowv[m] = e; ls += e;
  }
  red[tid] = ls; __syncthreads();
  for (int s = 128; s > 0; s >>= 1) {
    if (tid < s) red[tid] += red[tid + s];
    __syncthreads();
  }
  const float inv = 1.f / red[0];
  for (int m = tid; m < NN; m += 256)
    S[(size_t)row * NN + m] = (__bf16)(rowv[m] * inv);
}

// ---------------------------------------------------------------------------
// Transpose+pad shared weights to bf16 B-operand layout: Wt[d][k][o][i],
// i padded to CPAD with zeros.  Src layout: [d][k][i][o] (f32).
// ---------------------------------------------------------------------------
__global__ void k_wtrans(const float* __restrict__ W, __bf16* __restrict__ Wt,
                         int I, int O, int CPAD, int total) {
  int idx = blockIdx.x * 256 + threadIdx.x;
  if (idx >= total) return;
  int i  = idx % CPAD;
  int o  = (idx / CPAD) % O;
  int dk = idx / (CPAD * O);           // d*2+k
  float v = (i < I) ? W[((size_t)dk * I + i) * O + o] : 0.f;
  Wt[idx] = (__bf16)v;
}

// ---------------------------------------------------------------------------
// Pack inp = concat(x_t, state) -> bf16 in two layouts:
//   Xc[col][n]  (col = b*CPAD + c)    : B-operand of graph GEMM
//   Xa[n][b][c] (c padded to CPAD)    : A-operand of per-node GEMM
// Also pre-seeds the candidate buffers (x_t part + zero pads); the z*state
// part is filled later by the gate kernel.
// ---------------------------------------------------------------------------
__global__ void k_pack(const float* __restrict__ xsrc, int xmode, int t,
                       const float* __restrict__ state,
                       __bf16* __restrict__ Xc, __bf16* __restrict__ Xa,
                       __bf16* __restrict__ Cc, __bf16* __restrict__ Ca,
                       int INl, int C, int CPAD, int total) {
  int idx = blockIdx.x * 256 + threadIdx.x;
  if (idx >= total) return;
  int c = idx % CPAD;
  int b = (idx / CPAD) & (BB - 1);
  int n = idx / (CPAD * BB);
  float v;
  if (c < INl)
    v = (xmode == 0) ? xsrc[(size_t)(b * TT + t) * NN + n]          // source[b,t,n,0]
                     : xsrc[((size_t)n * BB + b) * HIDD + c];       // H1[n][b][c]
  else if (c < C)
    v = state[((size_t)n * BB + b) * HIDD + (c - INl)];
  else
    v = 0.f;
  __bf16 bv = (__bf16)v;
  Xa[idx] = bv;
  Xc[((size_t)(b * CPAD + c)) * NN + n] = bv;
  if (c < INl || c >= C) {             // x part + zero pads shared with cand
    Ca[idx] = bv;
    Cc[((size_t)(b * CPAD + c)) * NN + n] = bv;
  }
}

// ---------------------------------------------------------------------------
// Graph propagation GEMM:  Y[n][col] = sum_m S[n][m] * Xc[col][m]
// M=K=2048, COLS multiple of 128.  Block = 8 waves -> 128x128 tile,
// B operand staged through LDS, output stored bf16 in A-operand layout.
// ---------------------------------------------------------------------------
__global__ __launch_bounds__(256) void k_graph_gemm(const __bf16* __restrict__ S,
                                                    const __bf16* __restrict__ Xc,
                                                    __bf16* __restrict__ Y,
                                                    int COLS) {
  __shared__ __bf16 Bs[128 * 32];      // [col 0..127][k 0..31]
  const int tid   = threadIdx.x;
  const int wave  = tid >> 5, lane = tid & 31;
  const int lhalf = lane >> 4, l16 = lane & 15;
  const int colbase = blockIdx.x * 128;
  const int m0      = blockIdx.y * 128 + wave * 16;

  v8f acc[8];
#pragma unroll
  for (int ct = 0; ct < 8; ++ct)
#pragma unroll
    for (int j = 0; j < 8; ++j) acc[ct][j] = 0.f;

  const __bf16* arow = S + (size_t)(m0 + l16) * NN;
  const int fcol = tid >> 1, fpart = tid & 1;

  for (int kb = 0; kb < NN; kb += 32) {
    __syncthreads();
    {  // cooperative LDS fill of B tile (128 cols x 32 k)
      const __bf16* src = Xc + (size_t)(colbase + fcol) * NN + kb + fpart * 16;
      v8b u0 = *(const v8b*)src;
      v8b u1 = *(const v8b*)(src + 8);
      __bf16* dst = Bs + fcol * 32 + fpart * 16;
      *(v8b*)dst = u0;
      *(v8b*)(dst + 8) = u1;
    }
    __syncthreads();

    const __bf16* ap = arow + kb + lhalf * 8;
    v16b a = mk16(*(const v8b*)ap, *(const v8b*)(ap + 16));
    if (kb + 32 < NN) __builtin_prefetch(arow + kb + 32 + lhalf * 8, 0, 1);

#pragma unroll
    for (int ct = 0; ct < 8; ++ct) {
      const __bf16* bp = Bs + (ct * 16 + l16) * 32 + lhalf * 8;
      v16b bm = mk16(*(const v8b*)bp, *(const v8b*)(bp + 16));
      acc[ct] = wmma_bf16(a, bm, acc[ct]);
    }
  }
  // store D: row = m0 + j + lhalf*8, col = colbase + ct*16 + l16
#pragma unroll
  for (int ct = 0; ct < 8; ++ct)
#pragma unroll
    for (int j = 0; j < 8; ++j) {
      int row = m0 + j + lhalf * 8;
      int col = colbase + ct * 16 + l16;
      Y[(size_t)row * COLS + col] = (__bf16)acc[ct][j];
    }
}

// ---------------------------------------------------------------------------
// Gate GCN per node:  zr[b,o] = sigmoid( sum_d E[n,d] * (A_n . Wt_d) + bias )
// A_n = [x_g(k=0)|x_g(k=1)] 16 x (2*CPAD) bf16.  O = 128 -> 8 waves/node.
// z (o<64) -> cand = z*state written to both cand layouts; r (o>=64) -> Rbuf.
// ---------------------------------------------------------------------------
template <int CPAD>
__global__ __launch_bounds__(256) void k_node_gate(
    const __bf16* __restrict__ Xa, const __bf16* __restrict__ Ya,
    const __bf16* __restrict__ Wt, const float* __restrict__ bg,
    const float* __restrict__ E, const float* __restrict__ Hst,
    __bf16* __restrict__ Cc, __bf16* __restrict__ Ca,
    float* __restrict__ Rbuf, int INl) {
  constexpr int KT = CPAD / 32;
  const int n = blockIdx.x;
  const int wave = threadIdx.x >> 5, lane = threadIdx.x & 31;
  const int lhalf = lane >> 4, l16 = lane & 15;
  const int ocol = wave * 16 + l16;    // 0..127

  v16b a[2 * KT];
  {
    const __bf16* r0 = Xa + ((size_t)n * BB + l16) * CPAD;
    const __bf16* r1 = Ya + ((size_t)n * BB + l16) * CPAD;
#pragma unroll
    for (int k = 0; k < 2; ++k) {
      const __bf16* base = k ? r1 : r0;
#pragma unroll
      for (int ct = 0; ct < KT; ++ct) {
        const __bf16* p = base + ct * 32 + lhalf * 8;
        a[k * KT + ct] = mk16(*(const v8b*)p, *(const v8b*)(p + 16));
      }
    }
  }
  float e[ED];
  float bias = 0.f;
#pragma unroll
  for (int d = 0; d < ED; ++d) {
    e[d] = E[n * ED + d];
    bias += e[d] * bg[d * 128 + ocol];
  }
  v8f acc;
#pragma unroll
  for (int j = 0; j < 8; ++j) acc[j] = bias;

#pragma unroll
  for (int d = 0; d < ED; ++d) {
    v8f t;
#pragma unroll
    for (int j = 0; j < 8; ++j) t[j] = 0.f;
#pragma unroll
    for (int k = 0; k < 2; ++k) {
      const __bf16* wrow = Wt + ((size_t)(d * 2 + k) * 128 + ocol) * CPAD;
#pragma unroll
      for (int ct = 0; ct < KT; ++ct) {
        const __bf16* p = wrow + ct * 32 + lhalf * 8;
        v16b bm = mk16(*(const v8b*)p, *(const v8b*)(p + 16));
        t = wmma_bf16(a[k * KT + ct], bm, t);
      }
    }
#pragma unroll
    for (int j = 0; j < 8; ++j) acc[j] += e[d] * t[j];
  }
  // epilogue: sigmoid, split z|r
#pragma unroll
  for (int j = 0; j < 8; ++j) {
    int b = j + lhalf * 8;
    float sg = 1.f / (1.f + __expf(-acc[j]));
    if (ocol < HIDD) {
      float sv = Hst[((size_t)n * BB + b) * HIDD + ocol];
      float cv = sg * sv;                       // z * state
      __bf16 bv = (__bf16)cv;
      Ca[((size_t)n * BB + b) * CPAD + INl + ocol] = bv;
      Cc[((size_t)(b * CPAD + INl + ocol)) * NN + n] = bv;
    } else {
      Rbuf[((size_t)n * BB + b) * HIDD + (ocol - HIDD)] = sg;
    }
  }
}

// ---------------------------------------------------------------------------
// Candidate GCN + GRU update per node:  hc = tanh(...), h = r*h + (1-r)*hc
// O = 64 -> 4 waves/node.  Updates H in place.
// ---------------------------------------------------------------------------
template <int CPAD>
__global__ __launch_bounds__(128) void k_node_u(
    const __bf16* __restrict__ Ca, const __bf16* __restrict__ Yc,
    const __bf16* __restrict__ Wt, const float* __restrict__ bu,
    const float* __restrict__ E, const float* __restrict__ Rbuf,
    float* __restrict__ H) {
  constexpr int KT = CPAD / 32;
  const int n = blockIdx.x;
  const int wave = threadIdx.x >> 5, lane = threadIdx.x & 31;
  const int lhalf = lane >> 4, l16 = lane & 15;
  const int ocol = wave * 16 + l16;    // 0..63

  v16b a[2 * KT];
  {
    const __bf16* r0 = Ca + ((size_t)n * BB + l16) * CPAD;
    const __bf16* r1 = Yc + ((size_t)n * BB + l16) * CPAD;
#pragma unroll
    for (int k = 0; k < 2; ++k) {
      const __bf16* base = k ? r1 : r0;
#pragma unroll
      for (int ct = 0; ct < KT; ++ct) {
        const __bf16* p = base + ct * 32 + lhalf * 8;
        a[k * KT + ct] = mk16(*(const v8b*)p, *(const v8b*)(p + 16));
      }
    }
  }
  float e[ED];
  float bias = 0.f;
#pragma unroll
  for (int d = 0; d < ED; ++d) {
    e[d] = E[n * ED + d];
    bias += e[d] * bu[d * HIDD + ocol];
  }
  v8f acc;
#pragma unroll
  for (int j = 0; j < 8; ++j) acc[j] = bias;

#pragma unroll
  for (int d = 0; d < ED; ++d) {
    v8f t;
#pragma unroll
    for (int j = 0; j < 8; ++j) t[j] = 0.f;
#pragma unroll
    for (int k = 0; k < 2; ++k) {
      const __bf16* wrow = Wt + ((size_t)(d * 2 + k) * HIDD + ocol) * CPAD;
#pragma unroll
      for (int ct = 0; ct < KT; ++ct) {
        const __bf16* p = wrow + ct * 32 + lhalf * 8;
        v16b bm = mk16(*(const v8b*)p, *(const v8b*)(p + 16));
        t = wmma_bf16(a[k * KT + ct], bm, t);
      }
    }
#pragma unroll
    for (int j = 0; j < 8; ++j) acc[j] += e[d] * t[j];
  }
#pragma unroll
  for (int j = 0; j < 8; ++j) {
    int b = j + lhalf * 8;
    size_t hi = ((size_t)n * BB + b) * HIDD + ocol;
    float hc = tanhf(acc[j]);
    float r  = Rbuf[hi];
    H[hi] = r * H[hi] + (1.f - r) * hc;
  }
}

// ---------------------------------------------------------------------------
// out[b,h,n] = sum_c H2[n][b][c] * conv_w[h][c] + conv_b[h]
// ---------------------------------------------------------------------------
__global__ void k_out(const float* __restrict__ H2, const float* __restrict__ cw,
                      const float* __restrict__ cb, float* __restrict__ out) {
  int idx = blockIdx.x * 256 + threadIdx.x;     // B*HOR*N
  int n = idx % NN;
  int h = (idx / NN) % HORZ;
  int b = idx / (NN * HORZ);
  const float* hp = H2 + ((size_t)n * BB + b) * HIDD;
  const float* wp = cw + h * HIDD;
  float s = cb[h];
#pragma unroll
  for (int c = 0; c < HIDD; ++c) s += hp[c] * wp[c];
  out[idx] = s;
}

__global__ void k_zero(float* __restrict__ p, int total) {
  int i = blockIdx.x * 256 + threadIdx.x;
  if (i < total) p[i] = 0.f;
}

// ---------------------------------------------------------------------------
extern "C" void kernel_launch(void* const* d_in, const int* in_sizes, int n_in,
                              void* d_out, int out_size, void* d_ws, size_t ws_size,
                              hipStream_t stream) {
  const float* src = (const float*)d_in[0];
  const float* E   = (const float*)d_in[1];
  const float* Wg0 = (const float*)d_in[2];
  const float* bg0 = (const float*)d_in[3];
  const float* Wu0 = (const float*)d_in[4];
  const float* bu0 = (const float*)d_in[5];
  const float* Wg1 = (const float*)d_in[6];
  const float* bg1 = (const float*)d_in[7];
  const float* Wu1 = (const float*)d_in[8];
  const float* bu1 = (const float*)d_in[9];
  const float* cw  = (const float*)d_in[10];
  const float* cb  = (const float*)d_in[11];
  float* out = (float*)d_out;

  // scratch carve-out (bump allocator, 256B aligned)
  uint8_t* base = (uint8_t*)d_ws;
  size_t off = 0;
  auto alloc = [&](size_t bytes) -> void* {
    void* r = base + off;
    off = (off + bytes + 255) & ~(size_t)255;
    return r;
  };
  __bf16* S    = (__bf16*)alloc((size_t)NN * NN * 2);
  __bf16* Wt0g = (__bf16*)alloc((size_t)ED * 2 * 128 * 96 * 2);
  __bf16* Wt0u = (__bf16*)alloc((size_t)ED * 2 * 64 * 96 * 2);
  __bf16* Wt1g = (__bf16*)alloc((size_t)ED * 2 * 128 * 128 * 2);
  __bf16* Wt1u = (__bf16*)alloc((size_t)ED * 2 * 64 * 128 * 2);
  __bf16* Xc = (__bf16*)alloc((size_t)NN * NN * 2);
  __bf16* Xa = (__bf16*)alloc((size_t)NN * NN * 2);
  __bf16* Ya = (__bf16*)alloc((size_t)NN * NN * 2);
  __bf16* Cc = (__bf16*)alloc((size_t)NN * NN * 2);
  __bf16* Ca = (__bf16*)alloc((size_t)NN * NN * 2);
  __bf16* Yc = (__bf16*)alloc((size_t)NN * NN * 2);
  float* H1 = (float*)alloc((size_t)NN * BB * HIDD * 4);
  float* H2 = (float*)alloc((size_t)NN * BB * HIDD * 4);
  float* Rb = (float*)alloc((size_t)NN * BB * HIDD * 4);

  const int hsz = NN * BB * HIDD;
  k_zero<<<(hsz + 255) / 256, 256, 0, stream>>>(H1, hsz);
  k_zero<<<(hsz + 255) / 256, 256, 0, stream>>>(H2, hsz);

  // one-time precompute
  k_supports<<<NN, 256, 0, stream>>>(E, S);
  {
    int t0 = ED * 2 * 128 * 96;
    k_wtrans<<<(t0 + 255) / 256, 256, 0, stream>>>(Wg0, Wt0g, 65, 128, 96, t0);
    int t1 = ED * 2 * 64 * 96;
    k_wtrans<<<(t1 + 255) / 256, 256, 0, stream>>>(Wu0, Wt0u, 65, 64, 96, t1);
    int t2 = ED * 2 * 128 * 128;
    k_wtrans<<<(t2 + 255) / 256, 256, 0, stream>>>(Wg1, Wt1g, 128, 128, 128, t2);
    int t3 = ED * 2 * 64 * 128;
    k_wtrans<<<(t3 + 255) / 256, 256, 0, stream>>>(Wu1, Wt1u, 128, 64, 128, t3);
  }

  // layer configs: {INl, C, CPAD, COLS}
  const int IN0 = 1,  C0 = 65,  CP0 = 96,  COLS0 = BB * CP0;   // 1536
  const int IN1 = 64, C1 = 128, CP1 = 128, COLS1 = BB * CP1;   // 2048
  const int pk0 = NN * BB * CP0, pk1 = NN * BB * CP1;

  for (int t = 0; t < TT; ++t) {
    // ---- layer 0 (input = source, state = H1) ----
    k_pack<<<(pk0 + 255) / 256, 256, 0, stream>>>(src, 0, t, H1, Xc, Xa, Cc, Ca,
                                                  IN0, C0, CP0, pk0);
    k_graph_gemm<<<dim3(COLS0 / 128, NN / 128), 256, 0, stream>>>(S, Xc, Ya, COLS0);
    k_node_gate<96><<<NN, 256, 0, stream>>>(Xa, Ya, Wt0g, bg0, E, H1, Cc, Ca, Rb, IN0);
    k_graph_gemm<<<dim3(COLS0 / 128, NN / 128), 256, 0, stream>>>(S, Cc, Yc, COLS0);
    k_node_u<96><<<NN, 128, 0, stream>>>(Ca, Yc, Wt0u, bu0, E, Rb, H1);

    // ---- layer 1 (input = H1 (new), state = H2) ----
    k_pack<<<(pk1 + 255) / 256, 256, 0, stream>>>(H1, 1, t, H2, Xc, Xa, Cc, Ca,
                                                  IN1, C1, CP1, pk1);
    k_graph_gemm<<<dim3(COLS1 / 128, NN / 128), 256, 0, stream>>>(S, Xc, Ya, COLS1);
    k_node_gate<128><<<NN, 256, 0, stream>>>(Xa, Ya, Wt1g, bg1, E, H2, Cc, Ca, Rb, IN1);
    k_graph_gemm<<<dim3(COLS1 / 128, NN / 128), 256, 0, stream>>>(S, Cc, Yc, COLS1);
    k_node_u<128><<<NN, 128, 0, stream>>>(Ca, Yc, Wt1u, bu1, E, Rb, H2);
  }

  const int osz = BB * HORZ * NN;
  k_out<<<(osz + 255) / 256, 256, 0, stream>>>(H2, cw, cb, out);
}
